// InpaintGenerator_10866267259458
// MI455X (gfx1250) — compile-verified, matrix operand order checked by reading, not run on hardware
//
#include <hip/hip_runtime.h>
#include <math.h>

// ---------------------------------------------------------------------------
// Types / helpers
// ---------------------------------------------------------------------------
typedef unsigned short bfraw;                                   // raw bf16 bits
typedef __attribute__((ext_vector_type(16))) __bf16 v16bf;      // WMMA A/B frag
typedef __attribute__((ext_vector_type(8)))  float  v8f;        // WMMA C/D frag

struct alignas(16) U4 { unsigned x, y, z, w; };

union alignas(32) Frag16 { U4 q[2]; v16bf v; };

__device__ __forceinline__ float bf2f(bfraw h) {
    union { unsigned u; float f; } c; c.u = ((unsigned)h) << 16; return c.f;
}
__device__ __forceinline__ bfraw f2bf(float f) {
    union { float f; unsigned u; } c; c.f = f;
    unsigned u = c.u;
    return (bfraw)((u + 0x7FFFu + ((u >> 16) & 1u)) >> 16);   // RNE
}

#define BM 128
#define BN 64
#define LDT 40   // LDS row pitch in bf16 elements (80B: 16B-aligned, conflict-spread)

// ---------------------------------------------------------------------------
// Implicit-GEMM conv (NHWC bf16 activations, [O][K] bf16 weights, K=(ky*kw+kx)*Cin+ci)
// Requires: Cin % 32 == 0, Cout % 64 == 0, M = 16*Hout*Wout % 128 == 0.
// flags: 1 = leaky(0.2), 2 = add bf16 residual (same layout as out)
// ---------------------------------------------------------------------------
__global__ __launch_bounds__(256) void conv_wmma_kernel(
    const bfraw* __restrict__ src, const bfraw* __restrict__ wt,
    const float* __restrict__ bias, const bfraw* __restrict__ res,
    bfraw* __restrict__ out,
    int H, int W, int Cin, int Hout, int Wout, int Cout,
    int kh, int kw, int stride, int pad, int dil, int flags)
{
    __shared__ bfraw Als[BM * LDT];
    __shared__ bfraw Bls[BN * LDT];

    const int tid  = threadIdx.x;
    const int wid  = tid >> 5;
    const int lane = tid & 31;
    const int hh   = lane >> 4;
    const int r16  = lane & 15;
    const int m0   = blockIdx.x * BM;
    const int n0   = blockIdx.y * BN;
    const int K    = kh * kw * Cin;

    // A-stage role: 128 rows x 2 half-chunks of 16 bf16
    const int arow = tid >> 1, ahalf = tid & 1;
    const int p    = m0 + arow;
    const int HWo  = Hout * Wout;
    const int nimg = p / HWo;
    const int rem  = p - nimg * HWo;
    const int oy   = rem / Wout;
    const int ox   = rem - oy * Wout;
    const int ybase = oy * stride - pad;
    const int xbase = ox * stride - pad;

    // B-stage role: 64 rows x 4 chunks of 8 bf16
    const int brow = tid >> 2, bc = tid & 3;
    const size_t wbase = (size_t)(n0 + brow) * K + bc * 8;

    v8f acc[4];
#pragma unroll
    for (int s = 0; s < 4; ++s)
#pragma unroll
        for (int r = 0; r < 8; ++r) acc[s][r] = 0.f;

    const int ksteps = K >> 5;
    for (int kt = 0; kt < ksteps; ++kt) {
        const int k0 = kt << 5;

        // stage A (implicit im2col): within a 32-chunk (ky,kx) is constant
        {
            const int kk   = k0 + ahalf * 16;
            const int rker = kk / Cin;
            const int ci   = kk - rker * Cin;
            const int ky   = rker / kw;
            const int kx   = rker - ky * kw;
            const int y    = ybase + ky * dil;
            const int x    = xbase + kx * dil;
            U4 a0 = {0, 0, 0, 0}, a1 = {0, 0, 0, 0};
            if (y >= 0 && y < H && x >= 0 && x < W) {
                const U4* g = (const U4*)(src + ((((size_t)nimg * H + y) * W + x) * Cin + ci));
                a0 = g[0]; a1 = g[1];
            }
            *(U4*)(Als + arow * LDT + ahalf * 16)     = a0;
            *(U4*)(Als + arow * LDT + ahalf * 16 + 8) = a1;
        }
        // stage B (weights already [O][K] row-major == N-major fragment layout)
        *(U4*)(Bls + brow * LDT + bc * 8) = *(const U4*)(wt + wbase + k0);

        __syncthreads();

        Frag16 fa;
        const bfraw* ap = Als + (wid * 16 + r16) * LDT;
        fa.q[0] = *(const U4*)(ap + hh * 8);
        fa.q[1] = *(const U4*)(ap + 16 + hh * 8);
#pragma unroll
        for (int s = 0; s < 4; ++s) {
            Frag16 fb;
            const bfraw* bp = Bls + (s * 16 + r16) * LDT;
            fb.q[0] = *(const U4*)(bp + hh * 8);
            fb.q[1] = *(const U4*)(bp + 16 + hh * 8);
            acc[s] = __builtin_amdgcn_wmma_f32_16x16x32_bf16(
                false, fa.v, false, fb.v, (short)0, acc[s], false, false);
        }
        __syncthreads();
    }

    // epilogue: bias -> leaky -> +residual -> bf16 store
#pragma unroll
    for (int s = 0; s < 4; ++s) {
        const int gn  = n0 + s * 16 + r16;
        const float b = bias ? bias[gn] : 0.f;
#pragma unroll
        for (int r = 0; r < 8; ++r) {
            const int gm = m0 + wid * 16 + hh * 8 + r;
            float v = acc[s][r] + b;
            if (flags & 1) v = (v >= 0.f) ? v : 0.2f * v;
            const size_t o = (size_t)gm * Cout + gn;
            if (flags & 2) v += bf2f(res[o]);
            out[o] = f2bf(v);
        }
    }
}

// ---------------------------------------------------------------------------
// Plain GEMM: C[M][N] = A[M][K] * B[N][K]^T  (bf16 in; bf16 or f32 out)
// K % 32 == 0; M,N bounds-guarded.
// ---------------------------------------------------------------------------
__global__ __launch_bounds__(256) void gemm_wmma_kernel(
    const bfraw* __restrict__ A, const bfraw* __restrict__ B,
    bfraw* __restrict__ outb, float* __restrict__ outf,
    int M, int N, int Kd)
{
    __shared__ bfraw Als[BM * LDT];
    __shared__ bfraw Bls[BN * LDT];

    const int tid = threadIdx.x;
    const int wid = tid >> 5, lane = tid & 31;
    const int hh = lane >> 4, r16 = lane & 15;
    const int m0 = blockIdx.x * BM, n0 = blockIdx.y * BN;

    const int arow = tid >> 1, ahalf = tid & 1;
    const int gma  = m0 + arow;
    const bool aok = gma < M;
    const int brow = tid >> 2, bc = tid & 3;
    const int gnb  = n0 + brow;
    const bool bok = gnb < N;

    v8f acc[4];
#pragma unroll
    for (int s = 0; s < 4; ++s)
#pragma unroll
        for (int r = 0; r < 8; ++r) acc[s][r] = 0.f;

    const int ksteps = Kd >> 5;
    for (int kt = 0; kt < ksteps; ++kt) {
        const int k0 = kt << 5;
        U4 a0 = {0, 0, 0, 0}, a1 = {0, 0, 0, 0};
        if (aok) {
            const U4* g = (const U4*)(A + (size_t)gma * Kd + k0 + ahalf * 16);
            a0 = g[0]; a1 = g[1];
        }
        *(U4*)(Als + arow * LDT + ahalf * 16)     = a0;
        *(U4*)(Als + arow * LDT + ahalf * 16 + 8) = a1;

        U4 b0 = {0, 0, 0, 0};
        if (bok) b0 = *(const U4*)(B + (size_t)gnb * Kd + k0 + bc * 8);
        *(U4*)(Bls + brow * LDT + bc * 8) = b0;

        __syncthreads();

        Frag16 fa;
        const bfraw* ap = Als + (wid * 16 + r16) * LDT;
        fa.q[0] = *(const U4*)(ap + hh * 8);
        fa.q[1] = *(const U4*)(ap + 16 + hh * 8);
#pragma unroll
        for (int s = 0; s < 4; ++s) {
            Frag16 fb;
            const bfraw* bp = Bls + (s * 16 + r16) * LDT;
            fb.q[0] = *(const U4*)(bp + hh * 8);
            fb.q[1] = *(const U4*)(bp + 16 + hh * 8);
            acc[s] = __builtin_amdgcn_wmma_f32_16x16x32_bf16(
                false, fa.v, false, fb.v, (short)0, acc[s], false, false);
        }
        __syncthreads();
    }

#pragma unroll
    for (int s = 0; s < 4; ++s) {
        const int gn = n0 + s * 16 + r16;
#pragma unroll
        for (int r = 0; r < 8; ++r) {
            const int gm = m0 + wid * 16 + hh * 8 + r;
            if (gm < M && gn < N) {
                const size_t o = (size_t)gm * N + gn;
                if (outf) outf[o] = acc[s][r];
                else      outb[o] = f2bf(acc[s][r]);
            }
        }
    }
}

// ---------------------------------------------------------------------------
// Small support kernels
// ---------------------------------------------------------------------------
__global__ void repack_kernel(const float* __restrict__ w, bfraw* __restrict__ out,
                              int I, int KH, int KW, int total)
{
    int idx = blockIdx.x * 256 + threadIdx.x;
    if (idx >= total) return;
    const int ikk = I * KH * KW;
    const int o  = idx / ikk;
    const int r  = idx - o * ikk;
    const int ci = r / (KH * KW);
    const int rr = r - ci * (KH * KW);
    out[(size_t)o * ikk + rr * I + ci] = f2bf(w[idx]);
}

// encoder conv0: fp32 NCHW [16,3,120,640] -> bf16 NHWC [16,60,320,64], 3x3 s2 p1, leaky
__global__ void enc0_kernel(const float* __restrict__ in, const float* __restrict__ w,
                            const float* __restrict__ b, bfraw* __restrict__ out, int total)
{
    int idx = blockIdx.x * 256 + threadIdx.x;
    if (idx >= total) return;
    const int co = idx & 63;
    int t = idx >> 6;
    const int x = t % 320; t /= 320;
    const int y = t % 60;
    const int nimg = t / 60;
    float acc = b[co];
#pragma unroll
    for (int ky = 0; ky < 3; ++ky) {
        const int iy = 2 * y - 1 + ky;
        if (iy < 0 || iy >= 120) continue;
#pragma unroll
        for (int kx = 0; kx < 3; ++kx) {
            const int ix = 2 * x - 1 + kx;
            if (ix < 0 || ix >= 640) continue;
#pragma unroll
            for (int ci = 0; ci < 3; ++ci)
                acc += in[(((size_t)nimg * 3 + ci) * 120 + iy) * 640 + ix] *
                       w[((co * 3 + ci) * 3 + ky) * 3 + kx];
        }
    }
    acc = (acc >= 0.f) ? acc : 0.2f * acc;
    out[idx] = f2bf(acc);
}

// gather tokens: NHWC q/k (C=256 slice) -> [ntok][D] bf16
__global__ void gather_tok_kernel(const bfraw* __restrict__ src, bfraw* __restrict__ dst,
                                  int cbase, int oh, int ow, int ph, int pw,
                                  int H, int W, int D, int total)
{
    int idx = blockIdx.x * 256 + threadIdx.x;
    if (idx >= total) return;
    const int n = idx / D, d = idx - n * D;
    const int pp = ph * pw;
    const int ci = d / pp;
    const int r  = d - ci * pp;
    const int hy = r / pw, wx = r - (r / pw) * pw;
    const int ohow = oh * ow;
    const int tt = n / ohow;
    const int r2 = n - tt * ohow;
    const int oy = r2 / ow, ox = r2 - (r2 / ow) * ow;
    const int y = oy * ph + hy, x = ox * pw + wx;
    dst[idx] = src[(((size_t)tt * H + y) * W + x) * 256 + cbase + ci];
}

// gather V transposed + zero-pad: -> [D][mpad] bf16
__global__ void gather_vt_kernel(const bfraw* __restrict__ src, bfraw* __restrict__ dst,
                                 int cbase, int oh, int ow, int ph, int pw,
                                 int H, int W, int ntok, int mpad, int total)
{
    int idx = blockIdx.x * 256 + threadIdx.x;
    if (idx >= total) return;
    const int d = idx / mpad, j = idx - d * mpad;
    if (j >= ntok) { dst[idx] = 0; return; }
    const int pp = ph * pw;
    const int ci = d / pp;
    const int r  = d - ci * pp;
    const int hy = r / pw, wx = r - (r / pw) * pw;
    const int ohow = oh * ow;
    const int tt = j / ohow;
    const int r2 = j - tt * ohow;
    const int oy = r2 / ow, ox = r2 - (r2 / ow) * ow;
    const int y = oy * ph + hy, x = ox * pw + wx;
    dst[idx] = src[(((size_t)tt * H + y) * W + x) * 256 + cbase + ci];
}

// scatter tokens back: [ntok][D] -> NHWC (C=256 slice)
__global__ void scatter_tok_kernel(const bfraw* __restrict__ yt, bfraw* __restrict__ dst,
                                   int cbase, int oh, int ow, int ph, int pw,
                                   int H, int W, int D, int total)
{
    int idx = blockIdx.x * 256 + threadIdx.x;
    if (idx >= total) return;
    const int n = idx / D, d = idx - n * D;
    const int pp = ph * pw;
    const int ci = d / pp;
    const int r  = d - ci * pp;
    const int hy = r / pw, wx = r - (r / pw) * pw;
    const int ohow = oh * ow;
    const int tt = n / ohow;
    const int r2 = n - tt * ohow;
    const int oy = r2 / ow, ox = r2 - (r2 / ow) * ow;
    const int y = oy * ph + hy, x = ox * pw + wx;
    dst[(((size_t)tt * H + y) * W + x) * 256 + cbase + ci] = yt[idx];
}

// row softmax with scale; f32 scores [n][m] -> bf16 probs [n][mpad] (pad=0)
__global__ __launch_bounds__(256) void softmax_kernel(
    const float* __restrict__ sc, bfraw* __restrict__ pr, int m, int mpad, float scale)
{
    const int row = blockIdx.x;
    const float* s = sc + (size_t)row * m;
    __shared__ float red[256];
    const int tid = threadIdx.x;
    float mx = -3.4e38f;
    for (int j = tid; j < m; j += 256) mx = fmaxf(mx, s[j] * scale);
    red[tid] = mx; __syncthreads();
    for (int o = 128; o > 0; o >>= 1) { if (tid < o) red[tid] = fmaxf(red[tid], red[tid + o]); __syncthreads(); }
    mx = red[0]; __syncthreads();
    float sm = 0.f;
    for (int j = tid; j < m; j += 256) sm += __expf(s[j] * scale - mx);
    red[tid] = sm; __syncthreads();
    for (int o = 128; o > 0; o >>= 1) { if (tid < o) red[tid] += red[tid + o]; __syncthreads(); }
    const float inv = 1.f / red[0];
    bfraw* prow = pr + (size_t)row * mpad;
    for (int j = tid; j < m; j += 256) prow[j] = f2bf(__expf(s[j] * scale - mx) * inv);
    for (int j = m + tid; j < mpad; j += 256) prow[j] = 0;
}

// bilinear x2 upsample, align_corners=True, bf16 NHWC
__global__ void upsample_kernel(const bfraw* __restrict__ src, bfraw* __restrict__ dst,
                                int H, int W, int C, int total)
{
    int idx = blockIdx.x * 256 + threadIdx.x;
    if (idx >= total) return;
    const int c = idx % C;
    int t = idx / C;
    const int x2 = t % (2 * W); t /= (2 * W);
    const int y2 = t % (2 * H);
    const int nimg = t / (2 * H);
    const float cy = y2 * (float)(H - 1) / (float)(2 * H - 1);
    const float cx = x2 * (float)(W - 1) / (float)(2 * W - 1);
    const int y0 = (int)cy; const int y1 = min(y0 + 1, H - 1); const float wy = cy - y0;
    const int x0 = (int)cx; const int x1 = min(x0 + 1, W - 1); const float wx = cx - x0;
    const bfraw* b = src + (size_t)nimg * H * W * C;
    const float v00 = bf2f(b[((size_t)y0 * W + x0) * C + c]);
    const float v01 = bf2f(b[((size_t)y0 * W + x1) * C + c]);
    const float v10 = bf2f(b[((size_t)y1 * W + x0) * C + c]);
    const float v11 = bf2f(b[((size_t)y1 * W + x1) * C + c]);
    const float v = (v00 * (1.f - wx) + v01 * wx) * (1.f - wy) + (v10 * (1.f - wx) + v11 * wx) * wy;
    dst[idx] = f2bf(v);
}

// decoder final conv 64->3, p1, + tanh; bf16 NHWC in -> fp32 NCHW out
__global__ void d3_tanh_kernel(const bfraw* __restrict__ src, const float* __restrict__ w,
                               const float* __restrict__ b, float* __restrict__ out, int total)
{
    int idx = blockIdx.x * 256 + threadIdx.x;
    if (idx >= total) return;
    const int x = idx % 640;
    int t = idx / 640;
    const int y = t % 120; t /= 120;
    const int co = t % 3;
    const int nimg = t / 3;
    float acc = b[co];
    for (int ky = 0; ky < 3; ++ky) {
        const int iy = y - 1 + ky;
        if (iy < 0 || iy >= 120) continue;
        for (int kx = 0; kx < 3; ++kx) {
            const int ix = x - 1 + kx;
            if (ix < 0 || ix >= 640) continue;
            const bfraw* sp = src + (((size_t)nimg * 120 + iy) * 640 + ix) * 64;
            const float* wp = w + (size_t)co * 64 * 9 + ky * 3 + kx;
#pragma unroll 4
            for (int ci = 0; ci < 64; ++ci)
                acc += bf2f(sp[ci]) * wp[ci * 9];
        }
    }
    out[idx] = tanhf(acc);
}

// ---------------------------------------------------------------------------
// Host orchestration
// ---------------------------------------------------------------------------
extern "C" void kernel_launch(void* const* d_in, const int* in_sizes, int n_in,
                              void* d_out, int out_size, void* d_ws, size_t ws_size,
                              hipStream_t stream)
{
    const int NB = 16, Ht = 30, Wt = 160;      // transformer-resolution dims
    char* ws = (char*)d_ws;
    const size_t MB = (size_t)1 << 20;
    auto BFp = [&](size_t mb) -> bfraw* { return (bfraw*)(ws + mb * MB); };
    auto F32p = [&](size_t mb) -> float* { return (float*)(ws + mb * MB); };

    // pools (phase-overlapped)
    bfraw* E0 = BFp(0);   bfraw* E1 = BFp(40);  bfraw* E2 = BFp(80);
    bfraw* Qb = BFp(0);   bfraw* Kb = BFp(40);  bfraw* Vb = BFp(80);
    bfraw* AO = BFp(120); bfraw* FF1 = BFp(160);
    bfraw* QT = BFp(200); bfraw* KT = BFp(212); bfraw* VT = BFp(224); bfraw* YT = BFp(236);
    float* SC = F32p(248); bfraw* PR = BFp(356);
    bfraw* U0 = BFp(0);   bfraw* D0 = BFp(160); bfraw* D1 = BFp(240);
    bfraw* U1 = BFp(280); bfraw* D2 = BFp(440);
    bfraw* X  = BFp(600);
    bfraw* WR = BFp(640);

    // inputs
    const float* frames = (const float*)d_in[0];
    const float* enc_w0 = (const float*)d_in[1];  const float* enc_b0 = (const float*)d_in[2];
    const float* enc_w1 = (const float*)d_in[3];  const float* enc_b1 = (const float*)d_in[4];
    const float* enc_w2 = (const float*)d_in[5];  const float* enc_b2 = (const float*)d_in[6];
    const float* enc_w3 = (const float*)d_in[7];  const float* enc_b3 = (const float*)d_in[8];
    const float* tq_w = (const float*)d_in[9];   const float* tq_b = (const float*)d_in[10];
    const float* tk_w = (const float*)d_in[11];  const float* tk_b = (const float*)d_in[12];
    const float* tv_w = (const float*)d_in[13];  const float* tv_b = (const float*)d_in[14];
    const float* to_w = (const float*)d_in[15];  const float* to_b = (const float*)d_in[16];
    const float* tf1_w = (const float*)d_in[17]; const float* tf1_b = (const float*)d_in[18];
    const float* tf2_w = (const float*)d_in[19]; const float* tf2_b = (const float*)d_in[20];
    const float* d0_w = (const float*)d_in[21];  const float* d0_b = (const float*)d_in[22];
    const float* d1_w = (const float*)d_in[23];  const float* d1_b = (const float*)d_in[24];
    const float* d2_w = (const float*)d_in[25];  const float* d2_b = (const float*)d_in[26];
    const float* d3_w = (const float*)d_in[27];  const float* d3_b = (const float*)d_in[28];

    auto cdiv = [](int a, int b) { return (a + b - 1) / b; };
    auto repack = [&](const float* w, bfraw* dst, int O, int I, int KH, int KW) {
        const int total = O * I * KH * KW;
        repack_kernel<<<cdiv(total, 256), 256, 0, stream>>>(w, dst, I, KH, KW, total);
    };
    auto conv = [&](const bfraw* src, const bfraw* wt, const float* bias,
                    const bfraw* res, bfraw* out, int H, int W, int Cin,
                    int stride, int pad, int dil, int kh, int kw, int Cout, int flags) {
        const int Hout = (H + 2 * pad - (dil * (kh - 1) + 1)) / stride + 1;
        const int Wout = (W + 2 * pad - (dil * (kw - 1) + 1)) / stride + 1;
        const int M = NB * Hout * Wout;
        dim3 grid(M / BM, Cout / BN);
        conv_wmma_kernel<<<grid, 256, 0, stream>>>(src, wt, bias, res, out,
            H, W, Cin, Hout, Wout, Cout, kh, kw, stride, pad, dil, flags);
    };

    // ---- weight repack (bf16 [O][K]) ----
    const size_t oE1 = 0, oE2 = 36864, oE3 = 110592, oL = 405504;
    const size_t LS = 1966080;
    const size_t oQ = 0, oK = 65536, oV = 131072, oTO = 196608, oF1 = 786432, oF2 = 1376256;
    const size_t oD0 = 16134144, oD1 = 16429056, oD2 = 16502784;
    repack(enc_w1, WR + oE1, 64, 64, 3, 3);
    repack(enc_w2, WR + oE2, 128, 64, 3, 3);
    repack(enc_w3, WR + oE3, 256, 128, 3, 3);
    for (int i = 0; i < 8; ++i) {
        const size_t lw = oL + (size_t)i * LS;
        repack(tq_w + (size_t)i * 65536,  WR + lw + oQ,  256, 256, 1, 1);
        repack(tk_w + (size_t)i * 65536,  WR + lw + oK,  256, 256, 1, 1);
        repack(tv_w + (size_t)i * 65536,  WR + lw + oV,  256, 256, 1, 1);
        repack(to_w + (size_t)i * 589824, WR + lw + oTO, 256, 256, 3, 3);
        repack(tf1_w + (size_t)i * 589824, WR + lw + oF1, 256, 256, 3, 3);
        repack(tf2_w + (size_t)i * 589824, WR + lw + oF2, 256, 256, 3, 3);
    }
    repack(d0_w, WR + oD0, 128, 256, 3, 3);
    repack(d1_w, WR + oD1, 64, 128, 3, 3);
    repack(d2_w, WR + oD2, 64, 64, 3, 3);

    // ---- encoder ----
    {
        const int total = NB * 60 * 320 * 64;
        enc0_kernel<<<cdiv(total, 256), 256, 0, stream>>>(frames, enc_w0, enc_b0, E0, total);
    }
    conv(E0, WR + oE1, enc_b1, nullptr, E1, 60, 320, 64, 1, 1, 1, 3, 3, 64, 1);
    conv(E1, WR + oE2, enc_b2, nullptr, E2, 60, 320, 64, 2, 1, 1, 3, 3, 128, 1);
    conv(E2, WR + oE3, enc_b3, nullptr, X,  30, 160, 128, 1, 1, 1, 3, 3, 256, 1);

    // ---- transformer stack ----
    const int ohs[4] = {2, 5, 6, 10},  ows[4] = {2, 5, 16, 32};
    const int phs[4] = {15, 6, 5, 3},  pws[4] = {80, 32, 10, 5};
    const int ntoks[4] = {64, 400, 1536, 5120};
    const int Ds[4] = {76800, 12288, 3200, 960};
    const int mpads[4] = {64, 416, 1536, 5120};
    const int ntot = 4915200;  // ntok*D, identical for every patch group

    for (int i = 0; i < 8; ++i) {
        const size_t lw = oL + (size_t)i * LS;
        // q,k,v 1x1 convs (pure GEMMs through the implicit-conv path)
        conv(X, WR + lw + oQ, tq_b + i * 256, nullptr, Qb, Ht, Wt, 256, 1, 0, 1, 1, 1, 256, 0);
        conv(X, WR + lw + oK, tk_b + i * 256, nullptr, Kb, Ht, Wt, 256, 1, 0, 1, 1, 1, 256, 0);
        conv(X, WR + lw + oV, tv_b + i * 256, nullptr, Vb, Ht, Wt, 256, 1, 0, 1, 1, 1, 256, 0);

        for (int g = 0; g < 4; ++g) {
            const int cb = g * 64, oh = ohs[g], ow = ows[g], ph = phs[g], pw = pws[g];
            const int n = ntoks[g], D = Ds[g], mpad = mpads[g];
            gather_tok_kernel<<<cdiv(ntot, 256), 256, 0, stream>>>(Qb, QT, cb, oh, ow, ph, pw, Ht, Wt, D, ntot);
            gather_tok_kernel<<<cdiv(ntot, 256), 256, 0, stream>>>(Kb, KT, cb, oh, ow, ph, pw, Ht, Wt, D, ntot);
            const int vtot = D * mpad;
            gather_vt_kernel<<<cdiv(vtot, 256), 256, 0, stream>>>(Vb, VT, cb, oh, ow, ph, pw, Ht, Wt, n, mpad, vtot);
            // scores = Q K^T   (f32 out)
            {
                dim3 grid(cdiv(n, BM), cdiv(n, BN));
                gemm_wmma_kernel<<<grid, 256, 0, stream>>>(QT, KT, nullptr, SC, n, n, D);
            }
            softmax_kernel<<<n, 256, 0, stream>>>(SC, PR, n, mpad, 1.f / sqrtf((float)D));
            // y = P V  (bf16 out)
            {
                dim3 grid(cdiv(n, BM), D / BN);
                gemm_wmma_kernel<<<grid, 256, 0, stream>>>(PR, VT, YT, nullptr, n, D, mpad);
            }
            scatter_tok_kernel<<<cdiv(ntot, 256), 256, 0, stream>>>(YT, AO, cb, oh, ow, ph, pw, Ht, Wt, D, ntot);
        }
        // x = x + leaky(conv3x3(attn_out))
        conv(AO, WR + lw + oTO, to_b + i * 256, X, X, Ht, Wt, 256, 1, 1, 1, 3, 3, 256, 3);
        // feed-forward: x = x + leaky(conv3x3(leaky(conv3x3_dil2(x))))
        conv(X,  WR + lw + oF1, tf1_b + i * 256, nullptr, FF1, Ht, Wt, 256, 1, 2, 2, 3, 3, 256, 1);
        conv(FF1, WR + lw + oF2, tf2_b + i * 256, X, X, Ht, Wt, 256, 1, 1, 1, 3, 3, 256, 3);
    }

    // ---- decoder ----
    {
        const int total = NB * 60 * 320 * 256;
        upsample_kernel<<<cdiv(total, 256), 256, 0, stream>>>(X, U0, 30, 160, 256, total);
    }
    conv(U0, WR + oD0, d0_b, nullptr, D0, 60, 320, 256, 1, 1, 1, 3, 3, 128, 1);
    conv(D0, WR + oD1, d1_b, nullptr, D1, 60, 320, 128, 1, 1, 1, 3, 3, 64, 1);
    {
        const int total = NB * 120 * 640 * 64;
        upsample_kernel<<<cdiv(total, 256), 256, 0, stream>>>(D1, U1, 60, 320, 64, total);
    }
    conv(U1, WR + oD2, d2_b, nullptr, D2, 120, 640, 64, 1, 1, 1, 3, 3, 64, 1);
    {
        const int total = NB * 3 * 120 * 640;  // == out_size
        d3_tanh_kernel<<<cdiv(total, 256), 256, 0, stream>>>(D2, d3_w, d3_b, (float*)d_out, total);
    }
}
